// Decoder_3015067042231
// MI455X (gfx1250) — matrix operand order repaired and checked by reading, not compiled, not run on hardware
//
#include <hip/hip_runtime.h>
#include <hip/hip_bf16.h>
#include <math.h>
#include <stdint.h>

// ---------------------------------------------------------------------------
// Types for WMMA fragments (CDNA5 / gfx1250, wave32)
// ---------------------------------------------------------------------------
typedef __attribute__((ext_vector_type(16))) _Float16 v16h;
typedef __attribute__((ext_vector_type(8)))  _Float16 v8h;
typedef __attribute__((ext_vector_type(4)))  _Float16 v4h;
typedef __attribute__((ext_vector_type(8)))  float    v8f;
typedef __attribute__((ext_vector_type(4)))  unsigned int u32x4;
typedef __attribute__((ext_vector_type(8)))  unsigned int u32x8;

#define S_LEN 2048
#define E_DIM 1024
#define H_DIM 1024
#define V_DIM 32000
// TDM pad: pad_interval code 7 = 1024B, pad_amount code 3 = 16B
// -> LDS row stride = 1024 + 2*8 = 1040 halves; element K at K + 8*(K>=512)
#define TA_STRIDE 1040

// ---------------------------------------------------------------------------
// Helpers
// ---------------------------------------------------------------------------
__device__ __forceinline__ float dot4(const float4 a, const float4 b) {
  return a.x * b.x + a.y * b.y + a.z * b.z + a.w * b.w;
}

__device__ __forceinline__ float warp_reduce_sum(float v) {
#pragma unroll
  for (int off = 16; off > 0; off >>= 1)
    v += __shfl_down(v, off, 32);
  return v;
}

// Branchless transcendentals on the hardware TRANS pipe (v_exp_f32):
// these co-execute with the WMMA/XDL pipe instead of exec-mask branching.
__device__ __forceinline__ float fast_tanhf(float x) {
  const float e = __expf(2.0f * x);
  return 1.0f - 2.0f / (e + 1.0f);   // ->-1 for x<<0, ->1 for x>>0
}
__device__ __forceinline__ float fast_sigmoidf(float x) {
  return 1.0f / (1.0f + __expf(-x));
}

// ---------------------------------------------------------------------------
// K0a: W_att[:, H:H+E] (fp32, row stride 2048) -> f16 [1024 x 1024]
// grid: 1024 x 256
// ---------------------------------------------------------------------------
__global__ void conv_watt_kernel(const float* __restrict__ W_att,
                                 _Float16* __restrict__ wt16) {
  const int i   = blockIdx.x * blockDim.x + threadIdx.x;  // 0 .. 262143
  const int n   = (i * 4) >> 10;
  const int col = (i * 4) & 1023;
  const float4 f = *(const float4*)(W_att + (size_t)n * 2048 + 1024 + col);
  v4h hv = {(_Float16)f.x, (_Float16)f.y, (_Float16)f.z, (_Float16)f.w};
  *(v4h*)(wt16 + (size_t)n * 1024 + col) = hv;
}

// ---------------------------------------------------------------------------
// K0b: encoder_out (fp32 [2048x1024]) -> f16 [2048x1024] (TDM source tiles)
// grid: 2048 x 256
// ---------------------------------------------------------------------------
__global__ void conv_enc_kernel(const float* __restrict__ enc,
                                _Float16* __restrict__ enc16) {
  const int i   = blockIdx.x * blockDim.x + threadIdx.x;  // 0 .. 524287
  const int n   = (i * 4) >> 10;
  const int col = (i * 4) & 1023;
  const float4 f = *(const float4*)(enc + (size_t)n * 1024 + col);
  v4h hv = {(_Float16)f.x, (_Float16)f.y, (_Float16)f.z, (_Float16)f.w};
  *(v4h*)(enc16 + (size_t)n * 1024 + col) = hv;
}

// ---------------------------------------------------------------------------
// K1: gates = W_ih @ [last_ctx ; emb[word]] + b_ih + W_hh @ h0 + b_hh
// grid: 512 x 256 (wave32 per row, b128 streaming; 50 MB -> BW bound)
// ---------------------------------------------------------------------------
__global__ void gates_kernel(const float* __restrict__ W_ih,
                             const float* __restrict__ b_ih,
                             const float* __restrict__ W_hh,
                             const float* __restrict__ b_hh,
                             const float* __restrict__ last_ctx,
                             const float* __restrict__ emb,
                             const int*   __restrict__ word,
                             const float* __restrict__ h0,
                             float* __restrict__ gates) {
  const int wave = threadIdx.x >> 5, lane = threadIdx.x & 31;
  const int r = blockIdx.x * 8 + wave;
  const float4* w1 = (const float4*)(W_ih + (size_t)r * 2048);
  const float4* w2 = (const float4*)(W_hh + (size_t)r * 1024);
  const float4* lc = (const float4*)last_ctx;
  const float4* em = (const float4*)(emb + (size_t)word[0] * E_DIM);
  const float4* hp = (const float4*)h0;
  float acc = 0.0f;
  for (int j = lane; j < 256; j += 32) {
    acc += dot4(w1[j],       lc[j]);
    acc += dot4(w1[256 + j], em[j]);
    acc += dot4(w2[j],       hp[j]);
  }
  acc = warp_reduce_sum(acc);
  if (lane == 0) gates[r] = acc + b_ih[r] + b_hh[r];
}

// ---------------------------------------------------------------------------
// K2: LSTM cell (torch gate order i,f,g,o)
// ---------------------------------------------------------------------------
__global__ void lstm_kernel(const float* __restrict__ gates,
                            const float* __restrict__ c0,
                            float* __restrict__ h_ws,
                            float* __restrict__ out_h,
                            float* __restrict__ out_c) {
  for (int k = threadIdx.x; k < H_DIM; k += 256) {
    const float i = fast_sigmoidf(gates[k]);
    const float f = fast_sigmoidf(gates[H_DIM + k]);
    const float g = fast_tanhf(gates[2 * H_DIM + k]);
    const float o = fast_sigmoidf(gates[3 * H_DIM + k]);
    const float c = f * c0[k] + i * g;
    const float h = o * fast_tanhf(c);
    h_ws[k] = h; out_h[k] = h; out_c[k] = c;
  }
}

// ---------------------------------------------------------------------------
// K3: hpb[n] = W_att[n, 0:H] @ h + b_att[n]   grid: 128 x 256
// ---------------------------------------------------------------------------
__global__ void hproj_kernel(const float* __restrict__ W_att,
                             const float* __restrict__ b_att,
                             const float* __restrict__ h,
                             float* __restrict__ hpb) {
  const int wave = threadIdx.x >> 5, lane = threadIdx.x & 31;
  const int r = blockIdx.x * 8 + wave;
  const float4* w  = (const float4*)(W_att + (size_t)r * 2048);
  const float4* hv = (const float4*)h;
  float acc = 0.0f;
  for (int j = lane; j < 256; j += 32) acc += dot4(w[j], hv[j]);
  acc = warp_reduce_sum(acc);
  if (lane == 0) hpb[r] = acc + b_att[r];
}

// ---------------------------------------------------------------------------
// K4: attention scores via WMMA + TDM A-tile staging.
//   pre[s,n] = sum_k enc16[s,k] * wt16[n,k]      (f16 WMMA, f32 acc)
//   scores[s] = sum_n tanh(pre[s,n] + hpb[n]) * v[n]
//
// Staging: ONE tensor_load_to_lds per block (wave 0) copies the 16x2048B
// strip of enc16 into LDS. D# pad_enable inserts 16B after every 1024B so
// the LDS row stride is 1040 halves (bank-conflict padding done by the TDM).
// Completion via s_wait_tensorcnt 0, then workgroup barrier.
//
// Compute: K-outer / N-inner with 8 v8f accumulators per wave: one A-fragment
// ds_load feeds 8 back-to-back v_wmma_f32_16x16x32_f16.
// grid: 128 x 256
// ---------------------------------------------------------------------------
__global__ void attn_scores_kernel(const _Float16* __restrict__ enc16, // [2048,1024] f16
                                   const _Float16* __restrict__ wt16,  // [1024,1024] f16
                                   const float*   __restrict__ hpb,    // [1024]
                                   const float*   __restrict__ vvec,   // [1024]
                                   float*         __restrict__ scores) {
  __shared__ __align__(16) _Float16 tA[16 * TA_STRIDE];
  __shared__ float sc[16];
  const int tid = threadIdx.x;
  const int m0  = blockIdx.x * 16;

  // ---- Build the Tensor DMA Descriptor (D#) for this block's A tile ----
  // Element size 1B; tile = 2048B x 16 rows; tensor row stride 2048B.
  const unsigned long long ga =
      (unsigned long long)(size_t)(enc16 + (size_t)m0 * E_DIM);
  const unsigned lds_base = (unsigned)(size_t)(&tA[0]);  // LDS aperture: low 32b = LDS offset

  u32x4 g0;
  g0[0] = 1u;                                   // count=1, user descriptor
  g0[1] = lds_base;                             // lds_addr
  g0[2] = (unsigned)ga;                         // global_addr[31:0]
  g0[3] = (unsigned)(ga >> 32) | (2u << 30);    // global_addr[56:32] | type=2

  u32x8 g1;
  g1[0] = (1u << 20) | (7u << 22) | (3u << 25); // pad_enable | interval=1024B | amount=16B
  g1[1] = (2048u & 0xffffu) << 16;              // tensor_dim0[15:0]   (bits 63:48)
  g1[2] = (2048u >> 16) | ((2048u & 0xffffu) << 16); // tensor_dim0 hi | tensor_dim1 lo
  g1[3] = (2048u >> 16) | (2048u << 16);        // tensor_dim1 hi | tile_dim0=2048
  g1[4] = 16u;                                  // tile_dim1=16, tile_dim2=0
  g1[5] = 2048u;                                // tensor_dim0_stride[31:0]
  g1[6] = 0u;                                   // stride hi | dim1_stride lo
  g1[7] = 0u;

  if (tid < 32) {                               // wave 0 issues the DMA
    asm volatile("tensor_load_to_lds %0, %1" :: "s"(g0), "s"(g1) : "memory");
    __builtin_amdgcn_s_wait_tensorcnt(0);
  }
  if (tid < 16) sc[tid] = 0.0f;
  __syncthreads();

  const int wave = tid >> 5;
  const int lane = tid & 31;
  const int half = lane >> 4;
  const int l16  = lane & 15;

  // A fragment (16x32 f16): lane row = l16; 16 contiguous K at K=16*half.
  // B fragment (32x16 f16): lane col n; 16 contiguous K at K=16*half.
  const _Float16* aptr = &tA[l16 * TA_STRIDE + 16 * half];
  const _Float16* bbase = wt16 + (size_t)(wave * 128 + l16) * 1024 + 16 * half;

  v8f acc[8];
#pragma unroll
  for (int t = 0; t < 8; ++t) acc[t] = (v8f){};

#pragma unroll
  for (int kh = 0; kh < 2; ++kh) {              // padded halves of K (512 each)
    const _Float16* ap = aptr + kh * (512 + 8); // +8 halves TDM pad skip
    const _Float16* bp = bbase + kh * 512;
    for (int kt = 0; kt < 16; ++kt) {
      const v8h alo = *(const v8h*)(ap + kt * 32);
      const v8h ahi = *(const v8h*)(ap + kt * 32 + 8);
      const v16h a = __builtin_shufflevector(alo, ahi,
          0, 1, 2, 3, 4, 5, 6, 7, 8, 9, 10, 11, 12, 13, 14, 15);
#pragma unroll
      for (int t = 0; t < 8; ++t) {             // 8 WMMAs per A-fragment load
        const v16h b = *(const v16h*)(bp + (size_t)t * 16 * 1024 + kt * 32);
        acc[t] = __builtin_amdgcn_wmma_f32_16x16x32_f16(
            false, a, false, b, (short)0, acc[t], false, false);
      }
    }
  }

  // Fused epilogue: scores[m] += tanh(pre + hpb[n]) * v[n], reduced over N.
  float part[8];
#pragma unroll
  for (int r = 0; r < 8; ++r) part[r] = 0.0f;
#pragma unroll
  for (int t = 0; t < 8; ++t) {
    const int n = wave * 128 + t * 16 + l16;
    const float hb = hpb[n];
    const float vn = vvec[n];
#pragma unroll
    for (int r = 0; r < 8; ++r)                 // element (m = r+8*half, n)
      part[r] += fast_tanhf(acc[t][r] + hb) * vn;
  }
#pragma unroll
  for (int r = 0; r < 8; ++r)
    atomicAdd(&sc[r + 8 * half], part[r]);      // ds_add_f32 reduction over N
  __syncthreads();
  if (tid < 16) scores[m0 + tid] = sc[tid];
}

// ---------------------------------------------------------------------------
// K5a: softmax over scores[2048] -> wvec (+ d_out w)
// ---------------------------------------------------------------------------
__global__ void softmax_kernel(const float* __restrict__ scores,
                               float* __restrict__ wvec,
                               float* __restrict__ out_w) {
  __shared__ float red[256];
  const int tid = threadIdx.x;
  float m = -1e30f;
  for (int s = tid; s < S_LEN; s += 256) m = fmaxf(m, scores[s]);
  red[tid] = m; __syncthreads();
  for (int o = 128; o > 0; o >>= 1) {
    if (tid < o) red[tid] = fmaxf(red[tid], red[tid + o]);
    __syncthreads();
  }
  const float M = red[0];
  __syncthreads();
  float sum = 0.0f;
  for (int s = tid; s < S_LEN; s += 256) sum += __expf(scores[s] - M);
  red[tid] = sum; __syncthreads();
  for (int o = 128; o > 0; o >>= 1) {
    if (tid < o) red[tid] += red[tid + o];
    __syncthreads();
  }
  const float inv = 1.0f / red[0];
  for (int s = tid; s < S_LEN; s += 256) {
    const float w = __expf(scores[s] - M) * inv;
    wvec[s] = w; out_w[s] = w;
  }
}

// ---------------------------------------------------------------------------
// K5b: context[e] = sum_s w[s] * enc[s, e]    grid: 4 x 256
// ---------------------------------------------------------------------------
__global__ void context_kernel(const float* __restrict__ wvec,
                               const float* __restrict__ enc,
                               float* __restrict__ ctx) {
  const int e = blockIdx.x * 256 + threadIdx.x;
  float acc = 0.0f;
  for (int s = 0; s < S_LEN; ++s) acc += wvec[s] * enc[(size_t)s * E_DIM + e];
  ctx[e] = acc;
}

// ---------------------------------------------------------------------------
// K6: x_t = tanh(W_ah @ [context ; h] + b_ah)    grid: 128 x 256
// ---------------------------------------------------------------------------
__global__ void xt_kernel(const float* __restrict__ W_ah,
                          const float* __restrict__ b_ah,
                          const float* __restrict__ ctx,
                          const float* __restrict__ h,
                          float* __restrict__ xt_ws,
                          float* __restrict__ out_xt) {
  const int wave = threadIdx.x >> 5, lane = threadIdx.x & 31;
  const int r = blockIdx.x * 8 + wave;
  const float4* w  = (const float4*)(W_ah + (size_t)r * 2048);
  const float4* c4 = (const float4*)ctx;
  const float4* h4 = (const float4*)h;
  float acc = 0.0f;
  for (int j = lane; j < 256; j += 32) {
    acc += dot4(w[j], c4[j]);
    acc += dot4(w[256 + j], h4[j]);
  }
  acc = warp_reduce_sum(acc);
  if (lane == 0) {
    const float v = fast_tanhf(acc + b_ah[r]);
    xt_ws[r] = v; out_xt[r] = v;
  }
}

// ---------------------------------------------------------------------------
// K7: logits = W_out @ x_t + b_out   (131 MB stream -> HBM-bound tail)
// grid: 4000 x 256
// ---------------------------------------------------------------------------
__global__ void logits_kernel(const float* __restrict__ W_out,
                              const float* __restrict__ b_out,
                              const float* __restrict__ xt,
                              float* __restrict__ logits) {
  const int wave = threadIdx.x >> 5, lane = threadIdx.x & 31;
  const int r = blockIdx.x * 8 + wave;
  const float4* w  = (const float4*)(W_out + (size_t)r * 1024);
  const float4* x4 = (const float4*)xt;
  float acc = 0.0f;
  for (int j = lane; j < 256; j += 32) {
    __builtin_prefetch((const void*)(w + j + 64), 0, 0);  // global_prefetch
    acc += dot4(w[j], x4[j]);
  }
  acc = warp_reduce_sum(acc);
  if (lane == 0) logits[r] = acc + b_out[r];
}

// ---------------------------------------------------------------------------
// K8: in-place log_softmax over 32000 logits (single block)
// ---------------------------------------------------------------------------
__global__ void logsoftmax_kernel(float* __restrict__ out) {
  __shared__ float red[1024];
  const int tid = threadIdx.x;
  float m = -1e30f;
  for (int i = tid; i < V_DIM; i += 1024) m = fmaxf(m, out[i]);
  red[tid] = m; __syncthreads();
  for (int o = 512; o > 0; o >>= 1) {
    if (tid < o) red[tid] = fmaxf(red[tid], red[tid + o]);
    __syncthreads();
  }
  const float M = red[0];
  __syncthreads();
  float s = 0.0f;
  for (int i = tid; i < V_DIM; i += 1024) s += __expf(out[i] - M);
  red[tid] = s; __syncthreads();
  for (int o = 512; o > 0; o >>= 1) {
    if (tid < o) red[tid] += red[tid + o];
    __syncthreads();
  }
  const float lse = M + __logf(red[0]);
  for (int i = tid; i < V_DIM; i += 1024) out[i] = out[i] - lse;
}

// ---------------------------------------------------------------------------
// launch
// ---------------------------------------------------------------------------
extern "C" void kernel_launch(void* const* d_in, const int* in_sizes, int n_in,
                              void* d_out, int out_size, void* d_ws, size_t ws_size,
                              hipStream_t stream) {
  const float* enc   = (const float*)d_in[0];
  const int*   word  = (const int*)  d_in[1];
  const float* lctx  = (const float*)d_in[2];
  const float* h0    = (const float*)d_in[3];
  const float* c0    = (const float*)d_in[4];
  const float* emb   = (const float*)d_in[5];
  const float* W_ih  = (const float*)d_in[6];
  const float* b_ih  = (const float*)d_in[7];
  const float* W_hh  = (const float*)d_in[8];
  const float* b_hh  = (const float*)d_in[9];
  const float* W_att = (const float*)d_in[10];
  const float* b_att = (const float*)d_in[11];
  const float* vvec  = (const float*)d_in[12];
  const float* W_ah  = (const float*)d_in[13];
  const float* b_ah  = (const float*)d_in[14];
  const float* W_out = (const float*)d_in[15];
  const float* b_out = (const float*)d_in[16];

  // Output layout: [0,32000) log_softmax ; then h ; c ; x_t ; w
  float* out      = (float*)d_out;
  float* out_logp = out;
  float* out_h    = out + V_DIM;
  float* out_c    = out_h + H_DIM;
  float* out_xt   = out_c + H_DIM;
  float* out_w    = out_xt + H_DIM;

  // Workspace layout
  char* ws = (char*)d_ws;
  _Float16* wt16  = (_Float16*)ws;                        // 2 MB
  _Float16* enc16 = (_Float16*)(ws + 2u * 1024u * 1024u); // 4 MB
  float* gates   = (float*)(ws + 6u * 1024u * 1024u);     // 4096
  float* h_ws    = gates + 4096;                          // 1024
  float* hpb     = h_ws + 1024;                           // 1024
  float* scores  = hpb + 1024;                            // 2048
  float* wvec    = scores + 2048;                         // 2048
  float* ctx     = wvec + 2048;                           // 1024
  float* xt      = ctx + 1024;                            // 1024

  conv_watt_kernel  <<<1024, 256, 0, stream>>>(W_att, wt16);
  conv_enc_kernel   <<<2048, 256, 0, stream>>>(enc, enc16);
  gates_kernel      <<<512,  256, 0, stream>>>(W_ih, b_ih, W_hh, b_hh, lctx, emb, word, h0, gates);
  lstm_kernel       <<<1,    256, 0, stream>>>(gates, c0, h_ws, out_h, out_c);
  hproj_kernel      <<<128,  256, 0, stream>>>(W_att, b_att, h_ws, hpb);
  attn_scores_kernel<<<128,  256, 0, stream>>>(enc16, wt16, hpb, vvec, scores);
  softmax_kernel    <<<1,    256, 0, stream>>>(scores, wvec, out_w);
  context_kernel    <<<4,    256, 0, stream>>>(wvec, enc, ctx);
  xt_kernel         <<<128,  256, 0, stream>>>(W_ah, b_ah, ctx, h_ws, xt, out_xt);
  logits_kernel     <<<4000, 256, 0, stream>>>(W_out, b_out, xt, out_logp);
  logsoftmax_kernel <<<1,   1024, 0, stream>>>(out_logp);
}
